// Label_Branch_Without_Reconstruct_21612275434243
// MI455X (gfx1250) — matrix operand order repaired and checked by reading, not compiled, not run on hardware
//
#include <hip/hip_runtime.h>

// ---------------------------------------------------------------------------
// MI455X (gfx1250) implementation. wave32, WMMA f16 (f32 accum).
// Pipeline:
//   k0 stats_partial : per-(channel, batch-slice) sum/sumsq partials -> ws
//   k1 stats_final   : mean[c], rstd[c] -> ws
//   k2 branch_kernel : per (16-batch tile, domain): BN -> conv1 -> pool ->
//                      conv2 -> pool -> FC, all via v_wmma_f32_16x16x32_f16,
//                      branch outputs (D,B,L) f32 -> ws
//   k3 combine       : out[b,l] = sum_d weights[b,d] * branch[d,b,l]
// Deterministic (no atomics); ws usage ~1.6 MB.
// ---------------------------------------------------------------------------

typedef __attribute__((ext_vector_type(16))) _Float16 v16h;
typedef __attribute__((ext_vector_type(8)))  _Float16 v8h;
typedef __attribute__((ext_vector_type(8)))  float    v8f;

#define N_D 8
#define N_L 12
#define N_C 9
#define N_W 128
#define N_B 4096

static __device__ __forceinline__ v8f wmma16(v16h a, v16h b, v8f c) {
  // D = A(16x32 f16) * B(32x16 f16) + C(16x16 f32)
  return __builtin_amdgcn_wmma_f32_16x16x32_f16(false, a, false, b, (short)0, c,
                                                false, false);
}
static __device__ __forceinline__ v16h cat8(v8h lo, v8h hi) {
  return __builtin_shufflevector(lo, hi, 0, 1, 2, 3, 4, 5, 6, 7, 8, 9, 10, 11,
                                 12, 13, 14, 15);
}

// LDS: region 'a' is reused across phases (xn -> w2 fragments -> FC reduce).
struct __align__(32) Smem {
  union {
    _Float16 xn[130 * 16 * 16];    // [w(+2 guard)][b16][c16]  66560 B
    _Float16 w2f[9 * 4 * 32 * 16]; // [kp][nt][lane][h]        36864 B
    float    fcacc[4 * 16 * 16];   // [wave][b][n]              4096 B
  } a;
  _Float16 p1[60 * 16 * 32];       // [j][b][o]                61440 B
  _Float16 w1f[5 * 2 * 32 * 16];   // [kk][nt][lane][h]        10240 B
  _Float16 p2s[4][16 * 64];        // per-wave [b][o2]          8192 B
};                                  // total ~143 KB < 160 KB (CU mode)

// ---------------------------------------------------------------------------
__global__ __launch_bounds__(256) void stats_partial(const float* __restrict__ x,
                                                     float* __restrict__ part) {
  const int c = blockIdx.x >> 6;   // channel 0..8
  const int s = blockIdx.x & 63;   // batch slice of 64
  float s1 = 0.f, s2 = 0.f;
  for (int i = threadIdx.x; i < 64 * 128; i += 256) {
    const int b = s * 64 + (i >> 7);
    const int w = i & 127;
    const float v = x[((size_t)b * N_C + c) * N_W + w];
    s1 += v;
    s2 += v * v;
  }
  __shared__ float r1[256], r2[256];
  r1[threadIdx.x] = s1;
  r2[threadIdx.x] = s2;
  __syncthreads();
  for (int k = 128; k > 0; k >>= 1) {
    if (threadIdx.x < (unsigned)k) {
      r1[threadIdx.x] += r1[threadIdx.x + k];
      r2[threadIdx.x] += r2[threadIdx.x + k];
    }
    __syncthreads();
  }
  if (threadIdx.x == 0) {
    part[blockIdx.x]       = r1[0];
    part[576 + blockIdx.x] = r2[0];
  }
}

__global__ __launch_bounds__(32) void stats_final(const float* __restrict__ part,
                                                  float* __restrict__ mr) {
  const int c = threadIdx.x;
  if (c < N_C) {
    float s1 = 0.f, s2 = 0.f;
    for (int s = 0; s < 64; ++s) {
      s1 += part[c * 64 + s];
      s2 += part[576 + c * 64 + s];
    }
    const float inv = 1.f / ((float)N_B * (float)N_W);
    const float m   = s1 * inv;
    const float var = s2 * inv - m * m;
    mr[c]      = m;
    mr[16 + c] = rsqrtf(var + 1e-5f);
  }
}

// ---------------------------------------------------------------------------
__global__ __launch_bounds__(128) void branch_kernel(
    const float* __restrict__ x, const float* __restrict__ gamma,
    const float* __restrict__ beta, const float* __restrict__ w1,
    const float* __restrict__ b1, const float* __restrict__ w2,
    const float* __restrict__ b2, const float* __restrict__ fcw,
    const float* __restrict__ fcb, const float* __restrict__ mr,
    float* __restrict__ branch) {
  __shared__ Smem sm;
  const int tid    = threadIdx.x;
  const int wv     = tid >> 5;
  const int lane   = tid & 31;
  const int laneLo = lane & 15;
  const int hi8    = (lane >= 16) ? 8 : 0;   // A-fragment half-split
  const int hi16   = (lane >= 16) ? 16 : 0;  // B-fragment K-split
  const int d      = blockIdx.y;
  const int b0     = blockIdx.x * 16;

  // ---- stage 0: zero xn (covers c-pad and the 2 guard rows) ----------------
  for (int i = tid; i < 130 * 16 * 16; i += 128) sm.a.xn[i] = (_Float16)0.f;
  __syncthreads();

  // ---- stage 1: normalized input tile (f16) + conv1 B fragments ------------
  for (int i = tid; i < 16 * N_C * N_W; i += 128) {
    const int w = i & 127;
    const int c = (i >> 7) % 9;
    const int b = i / (9 * 128);
    const float xv = x[((size_t)(b0 + b) * N_C + c) * N_W + w];
    const float v  = (xv - mr[c]) * mr[16 + c] * gamma[d * N_C + c] +
                     beta[d * N_C + c];
    sm.a.xn[(w * 16 + b) * 16 + c] = (_Float16)v;
  }
  // w1 as B fragments; K-order = (tap k', c padded to 16), 5 chunks of 32.
  for (int i = tid; i < 5 * 2 * 32 * 16; i += 128) {
    const int h = i & 15, ls = (i >> 4) & 31, nt = (i >> 9) & 1, kk = i >> 10;
    const int kr = (h & 7) + ((ls >= 16) ? 8 : 0) + ((h >= 8) ? 16 : 0);
    const int kp = kk * 2 + (kr >> 4);
    const int c  = kr & 15;
    const int o  = nt * 16 + (ls & 15);
    const float v =
        (c < 9 && kp < 9) ? w1[(((size_t)d * 32 + o) * 9 + c) * 9 + kp] : 0.f;
    sm.w1f[i] = (_Float16)v;
  }
  __syncthreads();

  // ---- conv1: 120 t positions, waves split pooled outputs j = 0..59 --------
  for (int j = wv * 15; j < wv * 15 + 15; ++j) {
    v8f acc[2][2] = {};
#pragma unroll
    for (int tp = 0; tp < 2; ++tp) {
      const int t = 2 * j + tp;
#pragma unroll
      for (int kk = 0; kk < 5; ++kk) {
        v8h g0 = *(const v8h*)&sm.a.xn[((t + 2 * kk) * 16 + laneLo) * 16 + hi8];
        v8h g1 =
            *(const v8h*)&sm.a.xn[((t + 2 * kk + 1) * 16 + laneLo) * 16 + hi8];
        v16h afrag = cat8(g0, g1);
#pragma unroll
        for (int nt = 0; nt < 2; ++nt) {
          v16h bfrag = *(const v16h*)&sm.w1f[((kk * 2 + nt) * 32 + lane) * 16];
          acc[tp][nt] = wmma16(afrag, bfrag, acc[tp][nt]);
        }
      }
    }
#pragma unroll
    for (int nt = 0; nt < 2; ++nt) {
      const float bo = b1[d * 32 + nt * 16 + laneLo];
#pragma unroll
      for (int r = 0; r < 8; ++r) {
        const float v = fmaxf(fmaxf(acc[0][nt][r], acc[1][nt][r]) + bo, 0.f);
        const int bb  = r + hi8;
        sm.p1[(j * 16 + bb) * 32 + nt * 16 + laneLo] = (_Float16)v;
      }
    }
  }
  __syncthreads();  // p1 complete; xn dead -> reuse region for w2 fragments

  // ---- stage 2: conv2 B fragments; K-order = (tap, c2), 288 = 9x32 exact ---
  for (int i = tid; i < 9 * 4 * 32 * 16; i += 128) {
    const int h = i & 15, ls = (i >> 4) & 31, nt = (i >> 9) & 3, kp = i >> 11;
    const int c2 = (h & 7) + ((ls >= 16) ? 8 : 0) + ((h >= 8) ? 16 : 0);
    const int o2 = nt * 16 + (ls & 15);
    sm.a.w2f[i] = (_Float16)w2[(((size_t)d * 64 + o2) * 32 + c2) * 9 + kp];
  }
  __syncthreads();

  // ---- conv2 + fused FC: waves split pooled outputs j2 = 0..25 -------------
  v8f fcC = {};
  for (int j2 = wv; j2 < 26; j2 += 4) {
    v8f acc[2][4] = {};
#pragma unroll
    for (int tp = 0; tp < 2; ++tp) {
      const int t2 = 2 * j2 + tp;
#pragma unroll
      for (int kp = 0; kp < 9; ++kp) {
        const _Float16* row = &sm.p1[((t2 + kp) * 16 + laneLo) * 32];
        v8h g0 = *(const v8h*)&row[hi8];
        v8h g1 = *(const v8h*)&row[16 + hi8];
        v16h afrag = cat8(g0, g1);
#pragma unroll
        for (int nt = 0; nt < 4; ++nt) {
          v16h bfrag =
              *(const v16h*)&sm.a.w2f[((kp * 4 + nt) * 32 + lane) * 16];
          acc[tp][nt] = wmma16(afrag, bfrag, acc[tp][nt]);
        }
      }
    }
    // bias + relu + pool -> per-wave transpose buffer (A layout for FC)
#pragma unroll
    for (int nt = 0; nt < 4; ++nt) {
      const float bo = b2[d * 64 + nt * 16 + laneLo];
#pragma unroll
      for (int r = 0; r < 8; ++r) {
        const float v = fmaxf(fmaxf(acc[0][nt][r], acc[1][nt][r]) + bo, 0.f);
        const int bb  = r + hi8;
        sm.p2s[wv][bb * 64 + nt * 16 + laneLo] = (_Float16)v;
      }
    }
    // cross-lane LDS store->load within the wave: drain DS before reading
    asm volatile("s_wait_dscnt 0x0" ::: "memory");
#pragma unroll
    for (int kk = 0; kk < 2; ++kk) {
      const _Float16* prow = &sm.p2s[wv][laneLo * 64 + kk * 32];
      v8h g0 = *(const v8h*)&prow[hi8];
      v8h g1 = *(const v8h*)&prow[16 + hi8];
      v16h afrag = cat8(g0, g1);
      v16h bfrag;
#pragma unroll
      for (int h = 0; h < 16; ++h) {
        const int o2 = kk * 32 + (h & 7) + hi8 + ((h >= 8) ? 16 : 0);
        const float v =
            (laneLo < N_L)
                ? fcw[((size_t)d * N_L + laneLo) * 1664 + o2 * 26 + j2]
                : 0.f;
        bfrag[h] = (_Float16)v;
      }
      fcC = wmma16(afrag, bfrag, fcC);
    }
  }
  __syncthreads();  // w2f dead -> reuse region for cross-wave FC reduce

  // ---- FC reduce across the 4 waves, bias + relu, write branch output ------
#pragma unroll
  for (int r = 0; r < 8; ++r) {
    const int bb = r + hi8;
    sm.a.fcacc[(wv * 16 + bb) * 16 + laneLo] = fcC[r];
  }
  __syncthreads();
  if (tid < 16 * N_L) {
    const int bb = tid / N_L;
    const int l  = tid % N_L;
    float s = sm.a.fcacc[(0 * 16 + bb) * 16 + l] +
              sm.a.fcacc[(1 * 16 + bb) * 16 + l] +
              sm.a.fcacc[(2 * 16 + bb) * 16 + l] +
              sm.a.fcacc[(3 * 16 + bb) * 16 + l];
    s = fmaxf(s + fcb[d * N_L + l], 0.f);
    branch[((size_t)d * N_B + (b0 + bb)) * N_L + l] = s;
  }
}

// ---------------------------------------------------------------------------
__global__ __launch_bounds__(256) void combine_kernel(
    const float* __restrict__ wts, const float* __restrict__ branch,
    float* __restrict__ out) {
  const int i = blockIdx.x * 256 + threadIdx.x;
  if (i < N_B * N_L) {
    const int b = i / N_L;
    const int l = i % N_L;
    float s = 0.f;
#pragma unroll
    for (int dd = 0; dd < N_D; ++dd)
      s += wts[b * N_D + dd] * branch[((size_t)dd * N_B + b) * N_L + l];
    out[i] = s;
  }
}

// ---------------------------------------------------------------------------
extern "C" void kernel_launch(void* const* d_in, const int* in_sizes, int n_in,
                              void* d_out, int out_size, void* d_ws,
                              size_t ws_size, hipStream_t stream) {
  (void)in_sizes; (void)n_in; (void)out_size; (void)ws_size;
  const float* x     = (const float*)d_in[0];
  const float* wts   = (const float*)d_in[1];
  const float* gamma = (const float*)d_in[2];
  const float* beta  = (const float*)d_in[3];
  const float* w1    = (const float*)d_in[4];
  const float* b1    = (const float*)d_in[5];
  const float* w2    = (const float*)d_in[6];
  const float* b2    = (const float*)d_in[7];
  const float* fcw   = (const float*)d_in[8];
  const float* fcb   = (const float*)d_in[9];
  float* out = (float*)d_out;
  float* ws  = (float*)d_ws;

  float* part   = ws;          // 1152 f32 (sum + sumsq partials)
  float* mr     = ws + 1152;   // mean[9] @ +0, rstd[9] @ +16
  float* branch = ws + 1184;   // [8][4096][12] f32

  stats_partial<<<dim3(N_C * 64), 256, 0, stream>>>(x, part);
  stats_final<<<dim3(1), 32, 0, stream>>>(part, mr);
  branch_kernel<<<dim3(N_B / 16, N_D), 128, 0, stream>>>(
      x, gamma, beta, w1, b1, w2, b2, fcw, fcb, mr, branch);
  combine_kernel<<<dim3((N_B * N_L + 255) / 256), 256, 0, stream>>>(wts, branch,
                                                                    out);
}
